// RelativeSelfAttention_52037823758442
// MI455X (gfx1250) — matrix operand order repaired and checked by reading, not compiled, not run on hardware
//
#include <hip/hip_runtime.h>

#define SEQ 2048
#define DIM 1024
#define NH  16
#define DH  64
#define N3  3072

typedef __attribute__((ext_vector_type(16))) _Float16 v16h;
typedef __attribute__((ext_vector_type(8)))  _Float16 v8h;
typedef __attribute__((ext_vector_type(8)))  float    v8f;

static __device__ __forceinline__ v16h ld_frag(const _Float16* lo, const _Float16* hi) {
    v8h a = *(const v8h*)lo;
    v8h b = *(const v8h*)hi;
    v16h r;
#pragma unroll
    for (int i = 0; i < 8; ++i) { r[i] = a[i]; r[i + 8] = b[i]; }
    return r;
}

static __device__ __forceinline__ v8f wmma16(v16h a, v16h b, v8f c) {
    return __builtin_amdgcn_wmma_f32_16x16x32_f16(false, a, false, b, (short)0, c, false, false);
}

// ---------------- conversion kernels ----------------

__global__ void k_cvt_f16(const float* __restrict__ src, _Float16* __restrict__ dst, int n) {
    int i = blockIdx.x * blockDim.x + threadIdx.x;
    if (i < n) dst[i] = (_Float16)src[i];
}

// w: [rows][cols] f32 (row-major) -> wt: [cols][rows] f16
__global__ void k_transpose_cvt(const float* __restrict__ w, _Float16* __restrict__ wt,
                                int rows, int cols) {
    int i = blockIdx.x * blockDim.x + threadIdx.x;
    if (i >= rows * cols) return;
    int r = i / cols, c = i - r * cols;   // coalesced read of w
    wt[(size_t)c * rows + r] = (_Float16)w[i];
}

// ---------------- QKV GEMM (32x64 tile per wave) + rel-fold epilogue ----------------
// xh:[SEQ][DIM] f16, wt:[N3][DIM] f16 (Wqkv^T).
// Q/K stored [h][pos][dh]; V stored transposed [h][dh][pos]; K gets rel_emb[pos-1] added.
// Each 64-wide column tile is exactly one head of one part -> uniform epilogue branch.
__global__ void k_gemm_qkv(const _Float16* __restrict__ xh, const _Float16* __restrict__ wt,
                           const float* __restrict__ bqkv, const float* __restrict__ rel,
                           _Float16* __restrict__ Qh, _Float16* __restrict__ Kh,
                           _Float16* __restrict__ Vt) {
    const int NT = N3 / 64;                       // 48 column tiles
    int wave = (blockIdx.x * blockDim.x + threadIdx.x) >> 5;
    int lane = threadIdx.x & 31;
    int tm = wave / NT, tn = wave - tm * NT;      // tm in [0,64), tn in [0,48)
    int r = lane & 15, hf = lane >> 4;

    const _Float16* arow0 = xh + (size_t)(tm * 32 + r) * DIM;
    const _Float16* arow1 = arow0 + 16 * DIM;
    const _Float16* brow  = wt + (size_t)(tn * 64 + r) * DIM;

    v8f acc[2][4] = {};
#pragma unroll 2
    for (int k0 = 0; k0 < DIM; k0 += 32) {
        v16h a0 = ld_frag(arow0 + k0 + hf * 8, arow0 + k0 + 16 + hf * 8);
        v16h a1 = ld_frag(arow1 + k0 + hf * 8, arow1 + k0 + 16 + hf * 8);
#pragma unroll
        for (int ct = 0; ct < 4; ++ct) {
            const _Float16* bp = brow + (size_t)ct * 16 * DIM + k0 + hf * 8;
            v16h b = ld_frag(bp, bp + 16);
            acc[0][ct] = wmma16(a0, b, acc[0][ct]);
            acc[1][ct] = wmma16(a1, b, acc[1][ct]);
        }
    }

    int part = tn >> 4;        // 0=q 1=k 2=v  (uniform over wave)
    int h    = tn & 15;        // head          (uniform over wave)
#pragma unroll
    for (int ct = 0; ct < 4; ++ct) {
        int d = ct * 16 + r;
        float bias = bqkv[tn * 64 + d];
#pragma unroll
        for (int rt = 0; rt < 2; ++rt) {
#pragma unroll
            for (int v = 0; v < 8; ++v) {
                int i = tm * 32 + rt * 16 + hf * 8 + v;   // sequence position
                float val = acc[rt][ct][v] + bias;
                if (part == 0) {
                    Qh[((size_t)(h * SEQ + i)) * DH + d] = (_Float16)val;
                } else if (part == 1) {
                    if (i >= 1) val += rel[(size_t)(i - 1) * DH + d]; // skewed rel folded into K
                    Kh[((size_t)(h * SEQ + i)) * DH + d] = (_Float16)val;
                } else {
                    Vt[((size_t)(h * DH + d)) * SEQ + i] = (_Float16)val;
                }
            }
        }
    }
}

// ---------------- flash attention (no mask) ----------------
// One wave per (head, 16-query block). 8 waves / 256-thread block.
__global__ void k_attn(const _Float16* __restrict__ Qh, const _Float16* __restrict__ Kh,
                       const _Float16* __restrict__ Vt, _Float16* __restrict__ Of) {
    __shared__ __align__(32) _Float16 pbuf[8][16 * 32];
    int wib  = threadIdx.x >> 5;
    int wave = blockIdx.x * (blockDim.x >> 5) + wib;
    int lane = threadIdx.x & 31;
    int h  = wave >> 7;        // / (SEQ/16)
    int qb = wave & 127;
    int r = lane & 15, hf = lane >> 4;

    const _Float16* qrow = Qh + ((size_t)(h * SEQ + qb * 16 + r)) * DH;
    v16h aq0 = ld_frag(qrow +      hf * 8, qrow + 16 + hf * 8);   // dh 0..31
    v16h aq1 = ld_frag(qrow + 32 + hf * 8, qrow + 48 + hf * 8);   // dh 32..63

    float m[8], lsum[8];
#pragma unroll
    for (int v = 0; v < 8; ++v) { m[v] = -1e30f; lsum[v] = 0.f; }
    v8f o0 = {}, o1 = {}, o2 = {}, o3 = {};
    const float sc = 0.125f * 1.44269504088896f;   // (1/sqrt(64)) * log2(e)
    _Float16* pb = pbuf[wib];

    for (int kb = 0; kb < SEQ; kb += 32) {
        const _Float16* k0r = Kh + ((size_t)(h * SEQ + kb + r)) * DH;
        const _Float16* k1r = k0r + 16 * DH;
        v8f s0 = {}, s1 = {};
        {
            v16h b0 = ld_frag(k0r + hf * 8,      k0r + 16 + hf * 8);
            s0 = wmma16(aq0, b0, s0);
            v16h b1 = ld_frag(k0r + 32 + hf * 8, k0r + 48 + hf * 8);
            s0 = wmma16(aq1, b1, s0);
            v16h b2 = ld_frag(k1r + hf * 8,      k1r + 16 + hf * 8);
            s1 = wmma16(aq0, b2, s1);
            v16h b3 = ld_frag(k1r + 32 + hf * 8, k1r + 48 + hf * 8);
            s1 = wmma16(aq1, b3, s1);
        }
        float alpha[8];
#pragma unroll
        for (int v = 0; v < 8; ++v) {
            float t0 = s0[v] * sc, t1 = s1[v] * sc;
            float mx = fmaxf(t0, t1);
#pragma unroll
            for (int off = 1; off < 16; off <<= 1)
                mx = fmaxf(mx, __shfl_xor(mx, off, 32));   // row spans 16 lanes
            float mn = fmaxf(m[v], mx);
            float p0 = __builtin_exp2f(t0 - mn);
            float p1 = __builtin_exp2f(t1 - mn);
            float ps = p0 + p1;
#pragma unroll
            for (int off = 1; off < 16; off <<= 1)
                ps += __shfl_xor(ps, off, 32);
            alpha[v] = __builtin_exp2f(m[v] - mn);
            m[v] = mn;
            lsum[v] = lsum[v] * alpha[v] + ps;
            // stage P (C-layout -> row-major [16][32]) for A-fragment reload
            pb[(hf * 8 + v) * 32 + r]      = (_Float16)p0;
            pb[(hf * 8 + v) * 32 + 16 + r] = (_Float16)p1;
        }
        asm volatile("s_wait_dscnt 0" ::: "memory");
        v16h pa = ld_frag(pb + r * 32 + hf * 8, pb + r * 32 + 16 + hf * 8);
#pragma unroll
        for (int v = 0; v < 8; ++v) {
            o0[v] *= alpha[v]; o1[v] *= alpha[v];
            o2[v] *= alpha[v]; o3[v] *= alpha[v];
        }
        const _Float16* vbase = Vt + (size_t)(h * DH + r) * SEQ + kb + hf * 8;
        v16h vb;
        vb = ld_frag(vbase,            vbase + 16);            o0 = wmma16(pa, vb, o0);
        vb = ld_frag(vbase + 16 * SEQ, vbase + 16 * SEQ + 16); o1 = wmma16(pa, vb, o1);
        vb = ld_frag(vbase + 32 * SEQ, vbase + 32 * SEQ + 16); o2 = wmma16(pa, vb, o2);
        vb = ld_frag(vbase + 48 * SEQ, vbase + 48 * SEQ + 16); o3 = wmma16(pa, vb, o3);
    }
#pragma unroll
    for (int v = 0; v < 8; ++v) {
        int i = qb * 16 + hf * 8 + v;
        float inv = 1.f / lsum[v];
        size_t base = (size_t)i * DIM + h * DH;
        Of[base +  0 + r] = (_Float16)(o0[v] * inv);
        Of[base + 16 + r] = (_Float16)(o1[v] * inv);
        Of[base + 32 + r] = (_Float16)(o2[v] * inv);
        Of[base + 48 + r] = (_Float16)(o3[v] * inv);
    }
}

// ---------------- output projection (32x64 tile per wave) ----------------
__global__ void k_gemm_out(const _Float16* __restrict__ Of, const _Float16* __restrict__ wt,
                           const float* __restrict__ bout, float* __restrict__ out) {
    const int NT = DIM / 64;                      // 16 column tiles
    int wave = (blockIdx.x * blockDim.x + threadIdx.x) >> 5;
    int lane = threadIdx.x & 31;
    int tm = wave / NT, tn = wave - tm * NT;      // tm in [0,64), tn in [0,16)
    int r = lane & 15, hf = lane >> 4;

    const _Float16* arow0 = Of + (size_t)(tm * 32 + r) * DIM;
    const _Float16* arow1 = arow0 + 16 * DIM;
    const _Float16* brow  = wt + (size_t)(tn * 64 + r) * DIM;

    v8f acc[2][4] = {};
#pragma unroll 2
    for (int k0 = 0; k0 < DIM; k0 += 32) {
        v16h a0 = ld_frag(arow0 + k0 + hf * 8, arow0 + k0 + 16 + hf * 8);
        v16h a1 = ld_frag(arow1 + k0 + hf * 8, arow1 + k0 + 16 + hf * 8);
#pragma unroll
        for (int ct = 0; ct < 4; ++ct) {
            const _Float16* bp = brow + (size_t)ct * 16 * DIM + k0 + hf * 8;
            v16h b = ld_frag(bp, bp + 16);
            acc[0][ct] = wmma16(a0, b, acc[0][ct]);
            acc[1][ct] = wmma16(a1, b, acc[1][ct]);
        }
    }
#pragma unroll
    for (int ct = 0; ct < 4; ++ct) {
        int n = tn * 64 + ct * 16 + r;
        float bias = bout[n];
#pragma unroll
        for (int rt = 0; rt < 2; ++rt) {
#pragma unroll
            for (int v = 0; v < 8; ++v) {
                int i = tm * 32 + rt * 16 + hf * 8 + v;
                out[(size_t)i * DIM + n] = acc[rt][ct][v] + bias;
            }
        }
    }
}

extern "C" void kernel_launch(void* const* d_in, const int* in_sizes, int n_in,
                              void* d_out, int out_size, void* d_ws, size_t ws_size,
                              hipStream_t stream) {
    (void)in_sizes; (void)n_in; (void)out_size; (void)ws_size;
    const float* x    = (const float*)d_in[0];
    const float* Wqkv = (const float*)d_in[1];
    const float* bqkv = (const float*)d_in[2];
    const float* Wout = (const float*)d_in[3];
    const float* bout = (const float*)d_in[4];
    const float* rel  = (const float*)d_in[5];
    float* out = (float*)d_out;

    char* ws = (char*)d_ws;
    _Float16* xh  = (_Float16*)ws; ws += (size_t)SEQ * DIM * 2;     // 4 MB
    _Float16* WqT = (_Float16*)ws; ws += (size_t)N3  * DIM * 2;     // 6 MB
    _Float16* WoT = (_Float16*)ws; ws += (size_t)DIM * DIM * 2;     // 2 MB
    _Float16* Qh  = (_Float16*)ws; ws += (size_t)NH * SEQ * DH * 2; // 4 MB
    _Float16* Kh  = (_Float16*)ws; ws += (size_t)NH * SEQ * DH * 2; // 4 MB
    _Float16* Vt  = (_Float16*)ws; ws += (size_t)NH * DH * SEQ * 2; // 4 MB
    _Float16* Of  = (_Float16*)ws; ws += (size_t)SEQ * DIM * 2;     // 4 MB

    k_cvt_f16<<<(SEQ * DIM + 255) / 256, 256, 0, stream>>>(x, xh, SEQ * DIM);
    k_transpose_cvt<<<(DIM * N3 + 255) / 256, 256, 0, stream>>>(Wqkv, WqT, DIM, N3);
    k_transpose_cvt<<<(DIM * DIM + 255) / 256, 256, 0, stream>>>(Wout, WoT, DIM, DIM);

    // (SEQ/32)*(N3/64) = 64*48 = 3072 waves, 8 per block
    k_gemm_qkv<<<3072 / 8, 256, 0, stream>>>(xh, WqT, bqkv, rel, Qh, Kh, Vt);
    // NH*(SEQ/16) = 2048 waves
    k_attn<<<2048 / 8, 256, 0, stream>>>(Qh, Kh, Vt, Of);
    // (SEQ/32)*(DIM/64) = 64*16 = 1024 waves
    k_gemm_out<<<1024 / 8, 256, 0, stream>>>(Of, WoT, bout, out);
}
